// GlanceGazeAttention_73701638800171
// MI455X (gfx1250) — matrix operand order repaired and checked by reading, not compiled, not run on hardware
//
#include <hip/hip_runtime.h>

// ---------------------------------------------------------------------------
// GlanceGazeAttention for MI455X (gfx1250, wave32, WMMA 16x16x32 f16/f32-acc)
//
// Pipeline:
//   k_prep : qkv_w (192x576) -> qkvWt[576][192] f16 ; proj_w -> projWt[192][192] f16
//   k_attn : per (window b_, head h): x->LDS f16, q/k/v via WMMA (K=192),
//            S=q k^T via WMMA (K=32), softmax+rel-pos bias in LDS fp32,
//            O=P V via WMMA. Writes v (f16, for gaze) and xa (f16) to HBM.
//   k_gaze : per (image b, channel c): stage 72x72 haloed image from v via the
//            dilated-fold index map, 9x9 depthwise conv, accumulate into xa.
//   k_proj : (xa+g) @ proj_w + proj_b via WMMA -> fp32 out.
// ---------------------------------------------------------------------------

typedef __attribute__((ext_vector_type(16))) _Float16 v16h;
typedef __attribute__((ext_vector_type(8)))  _Float16 h8v;
typedef __attribute__((ext_vector_type(8)))  float    v8f;

#define NHEADS 6
#define HDIM   32
#define CDIM   192
#define NTOK   64
#define NWIN   64
#define NWH    8
#define WSZ    8
#define BIMG   32
#define BWIN   2048
#define GKS    9

// ---- WMMA fragment helpers (layouts per CDNA5 ISA 7.12.2) -------

// A fragment, 16x32 f16 tile from row-major matrix (ld in halves).
// Lane L: M = L%16; halves 0..7 -> K = 8*(L/16)+0..7 ; halves 8..15 -> K = 16+8*(L/16)+0..7
__device__ __forceinline__ v16h load_frag_a(const _Float16* base, int ld) {
  const int lane = threadIdx.x & 31;
  const int m = lane & 15, g = lane >> 4;
  const _Float16* r = base + m * ld + g * 8;
  h8v lo = *(const h8v*)(r);
  h8v hi = *(const h8v*)(r + 16);
  v16h out;
#pragma unroll
  for (int i = 0; i < 8; ++i) { out[i] = lo[i]; out[i + 8] = hi[i]; }
  return out;
}

// B fragment from a transposed operand Bt[N][K] (row-major, ld in halves).
// Lane L: N = L%16; halves 0..7 -> K = 8*(L/16)+0..7 ; halves 8..15 -> K = 16+8*(L/16)+0..7
// (same K order as the A fragment).
__device__ __forceinline__ v16h load_frag_b(const _Float16* baseT, int ld) {
  const int lane = threadIdx.x & 31;
  const int n = lane & 15, g = lane >> 4;
  const _Float16* r = baseT + n * ld + g * 8;
  h8v lo = *(const h8v*)(r);
  h8v hi = *(const h8v*)(r + 16);
  v16h out;
#pragma unroll
  for (int i = 0; i < 8; ++i) { out[i] = lo[i]; out[i + 8] = hi[i]; }
  return out;
}

__device__ __forceinline__ v8f wmma32(v16h a, v16h b, v8f c) {
  return __builtin_amdgcn_wmma_f32_16x16x32_f16(false, a, false, b, (short)0, c,
                                                false, false);
}

// C/D layout: lane L: N = L%16; VGPR r -> M = 8*(L/16) + r
// ---------------------------------------------------------------------------

__global__ void k_prep(const float* __restrict__ qkv_w,
                       const float* __restrict__ proj_w,
                       _Float16* __restrict__ qkvWt,
                       _Float16* __restrict__ projWt) {
  int i = blockIdx.x * blockDim.x + threadIdx.x;
  if (i < 576 * 192) {
    int n = i / 192, k = i % 192;
    qkvWt[i] = (_Float16)qkv_w[k * 576 + n];
  }
  if (i < 192 * 192) {
    int n = i / 192, k = i % 192;
    projWt[i] = (_Float16)proj_w[k * 192 + n];
  }
}

__global__ void __launch_bounds__(128)
k_attn(const float* __restrict__ x, const _Float16* __restrict__ qkvWt,
       const float* __restrict__ qkv_b, const float* __restrict__ rpb,
       _Float16* __restrict__ vh, _Float16* __restrict__ xa) {
  __shared__ __attribute__((aligned(32))) _Float16 sx[NTOK * CDIM];   // 24 KB
  __shared__ __attribute__((aligned(32))) _Float16 sq[NTOK * HDIM];   // 4 KB
  __shared__ __attribute__((aligned(32))) _Float16 sk[NTOK * HDIM];   // 4 KB (= k^T operand)
  __shared__ __attribute__((aligned(32))) _Float16 svT[HDIM * NTOK];  // 4 KB (v transposed)
  __shared__ __attribute__((aligned(16))) float    sS[NTOK * NTOK];   // 16 KB
  __shared__ __attribute__((aligned(32))) _Float16 sP[NTOK * NTOK];   // 8 KB

  const int h = blockIdx.x;          // head
  const int b_ = blockIdx.y;         // window index
  const int tid = threadIdx.x;
  const int wave = tid >> 5;
  const int ln = tid & 15;           // N within tile for B/C fragments
  const int lg = (tid & 31) >> 4;    // lane group

  // ---- phase 1: stage x window as f16 ----
  const float4* xg = (const float4*)(x + (size_t)b_ * NTOK * CDIM);
  for (int i = tid; i < NTOK * CDIM / 4; i += 128) {
    float4 v4 = xg[i];
    int o = i * 4;
    sx[o + 0] = (_Float16)v4.x; sx[o + 1] = (_Float16)v4.y;
    sx[o + 2] = (_Float16)v4.z; sx[o + 3] = (_Float16)v4.w;
  }
  __syncthreads();

  // ---- phase 2: q,k,v = x @ W (per-head 64x32 slabs), wave w -> rows 16w..16w+15
  const float scale = 0.17677669529663687f;  // 32^-0.5
  _Float16* vg = vh + (((size_t)b_ * NHEADS + h) * NTOK) * HDIM;
  for (int mat = 0; mat < 3; ++mat) {        // 0=q 1=k 2=v
    for (int ct = 0; ct < 2; ++ct) {         // 16-wide column tiles of head dim
      v8f acc = {};
      const _Float16* bt = qkvWt + (size_t)(mat * CDIM + h * HDIM + ct * 16) * CDIM;
#pragma unroll
      for (int kk = 0; kk < 6; ++kk) {
        v16h a = load_frag_a(sx + wave * 16 * CDIM + kk * 32, CDIM);
        v16h b = load_frag_b(bt + kk * 32, CDIM);
        acc = wmma32(a, b, acc);
      }
      float bias = qkv_b[mat * CDIM + h * HDIM + ct * 16 + ln];
#pragma unroll
      for (int r = 0; r < 8; ++r) acc[r] += bias;
      if (mat == 0) {
#pragma unroll
        for (int r = 0; r < 8; ++r) acc[r] *= scale;
#pragma unroll
        for (int r = 0; r < 8; ++r)
          sq[(wave * 16 + lg * 8 + r) * HDIM + ct * 16 + ln] = (_Float16)acc[r];
      } else if (mat == 1) {
#pragma unroll
        for (int r = 0; r < 8; ++r)
          sk[(wave * 16 + lg * 8 + r) * HDIM + ct * 16 + ln] = (_Float16)acc[r];
      } else {
#pragma unroll
        for (int r = 0; r < 8; ++r) {
          _Float16 hv = (_Float16)acc[r];
          svT[(ct * 16 + ln) * NTOK + wave * 16 + lg * 8 + r] = hv;  // v^T for P@V
          vg[(wave * 16 + lg * 8 + r) * HDIM + ct * 16 + ln] = hv;   // v for gaze
        }
      }
    }
  }
  __syncthreads();

  // ---- phase 3: S = (q*scale) @ k^T ; one WMMA per 16x16 tile (K=32) ----
  {
    v16h a = load_frag_a(sq + wave * 16 * HDIM, HDIM);
    for (int jt = 0; jt < 4; ++jt) {
      v16h b = load_frag_b(sk + jt * 16 * HDIM, HDIM);
      v8f c = {};
      c = wmma32(a, b, c);
#pragma unroll
      for (int r = 0; r < 8; ++r)
        sS[(wave * 16 + lg * 8 + r) * NTOK + jt * 16 + ln] = c[r];
    }
  }
  __syncthreads();

  // ---- phase 4: relative-position bias + softmax (row per thread) ----
  if (tid < NTOK) {
    const int nrow = tid;
    const int pn = nrow >> 3, qn = nrow & 7;
    float mx = -1e30f;
    for (int m = 0; m < NTOK; ++m) {
      int pm = m >> 3, qm = m & 7;
      int ridx = (pn - pm + 7) * 15 + (qn - qm + 7);
      float v = sS[nrow * NTOK + m] + rpb[ridx * NHEADS + h];
      sS[nrow * NTOK + m] = v;
      mx = fmaxf(mx, v);
    }
    float sum = 0.f;
    for (int m = 0; m < NTOK; ++m) {
      float e = __expf(sS[nrow * NTOK + m] - mx);
      sS[nrow * NTOK + m] = e;
      sum += e;
    }
    float rinv = 1.f / sum;
    for (int m = 0; m < NTOK; ++m)
      sP[nrow * NTOK + m] = (_Float16)(sS[nrow * NTOK + m] * rinv);
  }
  __syncthreads();

  // ---- phase 5: O = P @ v ; write xa (channel = h*32+d) ----
  {
    v16h a0 = load_frag_a(sP + wave * 16 * NTOK + 0, NTOK);
    v16h a1 = load_frag_a(sP + wave * 16 * NTOK + 32, NTOK);
    for (int dt = 0; dt < 2; ++dt) {
      v8f c = {};
      c = wmma32(a0, load_frag_b(svT + (dt * 16) * NTOK + 0, NTOK), c);
      c = wmma32(a1, load_frag_b(svT + (dt * 16) * NTOK + 32, NTOK), c);
      _Float16* og = xa + ((size_t)b_ * NTOK) * CDIM + h * HDIM + dt * 16;
#pragma unroll
      for (int r = 0; r < 8; ++r)
        og[(size_t)(wave * 16 + lg * 8 + r) * CDIM + ln] = (_Float16)c[r];
    }
  }
}

// Dilated fold -> 9x9 depthwise conv -> unfold, accumulated into xa.
// Image pixel (y,x): y = p*8+i0, x = q*8+j0 ; window w = i0*8+j0, token n = p*8+q.
__global__ void __launch_bounds__(256)
k_gaze(const _Float16* __restrict__ vh, const float* __restrict__ gw,
       const float* __restrict__ gb, _Float16* __restrict__ xa) {
  __shared__ _Float16 simg[72 * 72];
  __shared__ float swt[GKS * GKS];
  const int c = blockIdx.x;  // channel
  const int b = blockIdx.y;  // image
  const int h = c >> 5, d = c & 31;
  const int tid = threadIdx.x;

  if (tid < GKS * GKS) swt[tid] = gw[c * GKS * GKS + tid];
  for (int i = tid; i < 72 * 72; i += 256) {
    int yy = i / 72 - 4, xx = i % 72 - 4;
    _Float16 val = (_Float16)0.f;
    if (yy >= 0 && yy < 64 && xx >= 0 && xx < 64) {
      int p = yy >> 3, i0 = yy & 7, q = xx >> 3, j0 = xx & 7;
      size_t bw = (size_t)b * NWIN + i0 * 8 + j0;
      int n = p * 8 + q;
      val = vh[((bw * NHEADS + h) * NTOK + n) * HDIM + d];
    }
    simg[i] = val;
  }
  __syncthreads();

  const float gbias = gb[c];
  for (int pix = tid; pix < 64 * 64; pix += 256) {
    int y = pix >> 6, xx = pix & 63;
    float acc = gbias;
#pragma unroll
    for (int ky = 0; ky < GKS; ++ky)
#pragma unroll
      for (int kx = 0; kx < GKS; ++kx)
        acc += (float)simg[(y + ky) * 72 + (xx + kx)] * swt[ky * GKS + kx];
    int p = y >> 3, i0 = y & 7, q = xx >> 3, j0 = xx & 7;
    size_t bw = (size_t)b * NWIN + i0 * 8 + j0;
    size_t idx = (bw * NTOK + (p * 8 + q)) * CDIM + c;
    xa[idx] = (_Float16)((float)xa[idx] + acc);
  }
}

__global__ void __launch_bounds__(128)
k_proj(const _Float16* __restrict__ xa, const _Float16* __restrict__ projWt,
       const float* __restrict__ proj_b, float* __restrict__ out) {
  __shared__ __attribute__((aligned(32))) _Float16 sA[64 * CDIM];  // 24 KB
  const int tid = threadIdx.x, wave = tid >> 5;
  const int ln = tid & 15, lg = (tid & 31) >> 4;
  const size_t r0 = (size_t)blockIdx.x * 64;

  const uint4* src = (const uint4*)(xa + r0 * CDIM);
  uint4* dst = (uint4*)sA;
  for (int i = tid; i < 64 * CDIM * 2 / 16; i += 128) dst[i] = src[i];
  __syncthreads();

  v16h a[6];
#pragma unroll
  for (int kk = 0; kk < 6; ++kk)
    a[kk] = load_frag_a(sA + wave * 16 * CDIM + kk * 32, CDIM);

  for (int ct = 0; ct < 12; ++ct) {
    v8f acc = {};
    const _Float16* bt = projWt + (size_t)(ct * 16) * CDIM;
#pragma unroll
    for (int kk = 0; kk < 6; ++kk)
      acc = wmma32(a[kk], load_frag_b(bt + kk * 32, CDIM), acc);
    float pb = proj_b[ct * 16 + ln];
    float* og = out + (r0 + wave * 16) * CDIM + ct * 16;
#pragma unroll
    for (int r = 0; r < 8; ++r)
      og[(size_t)(lg * 8 + r) * CDIM + ln] = acc[r] + pb;
  }
}

extern "C" void kernel_launch(void* const* d_in, const int* in_sizes, int n_in,
                              void* d_out, int out_size, void* d_ws, size_t ws_size,
                              hipStream_t stream) {
  (void)in_sizes; (void)n_in; (void)out_size; (void)ws_size;
  const float* x      = (const float*)d_in[0];
  const float* qkv_w  = (const float*)d_in[1];
  const float* qkv_b  = (const float*)d_in[2];
  const float* proj_w = (const float*)d_in[3];
  const float* proj_b = (const float*)d_in[4];
  const float* rpb    = (const float*)d_in[5];
  const float* gaze_w = (const float*)d_in[6];
  const float* gaze_b = (const float*)d_in[7];
  float* out = (float*)d_out;

  char* ws = (char*)d_ws;
  _Float16* qkvWt  = (_Float16*)(ws);                       // 576*192*2   = 221184 B
  _Float16* projWt = (_Float16*)(ws + 221184);              // 192*192*2   = 73728 B
  _Float16* vh     = (_Float16*)(ws + 294912);              // 2048*6*64*32*2 = 50331648 B
  _Float16* xa     = (_Float16*)(ws + 50626560);            // 2048*64*192*2  = 50331648 B
  // total ws: 100,958,208 bytes

  k_prep<<<(576 * 192 + 255) / 256, 256, 0, stream>>>(qkv_w, proj_w, qkvWt, projWt);
  k_attn<<<dim3(NHEADS, BWIN), 128, 0, stream>>>(x, qkvWt, qkv_b, rpb, vh, xa);
  k_gaze<<<dim3(CDIM, BIMG), 256, 0, stream>>>(vh, gaze_w, gaze_b, xa);
  k_proj<<<BWIN * NTOK / 64, 128, 0, stream>>>(xa, projWt, proj_b, out);
}
